// SRUCell_86517821214226
// MI455X (gfx1250) — matrix operand order repaired and checked by reading, not compiled, not run on hardware
//
#include <hip/hip_runtime.h>
#include <stdint.h>

// ---------------------------------------------------------------------------
// SRU layer for MI455X (gfx1250, wave32).
//   Phase 1: U = X @ W  via split-bf16 WMMA (v_wmma_f32_16x16x32_bf16)
//            a*b ~= aHi*bHi + aHi*bLo + aLo*bHi  (fp32-class accuracy)
//   Phase 2: sequential SRU scan over L, parallel over B*D.
// Workspace: U (65536 x 1536 fp32 = 403 MB) in d_ws.
// ---------------------------------------------------------------------------

typedef __attribute__((ext_vector_type(16))) __bf16        v16bf;
typedef __attribute__((ext_vector_type(8)))  float         v8f;
typedef __attribute__((ext_vector_type(4)))  unsigned int  uint4v;

union FragBF {
    v16bf  bf;
    uint4v q[2];
};
union F4 {
    float4 v;
    float  f[4];
};

__device__ __forceinline__ unsigned short f32_to_bf16_rne(float f) {
    unsigned int u = __float_as_uint(f);
    u += 0x7FFFu + ((u >> 16) & 1u);            // round-to-nearest-even
    return (unsigned short)(u >> 16);
}
__device__ __forceinline__ float bf16_as_f32(unsigned short h) {
    return __uint_as_float(((unsigned int)h) << 16);
}
__device__ __forceinline__ void bf16_split(float f, unsigned short& hi, unsigned short& lo) {
    hi = f32_to_bf16_rne(f);
    lo = f32_to_bf16_rne(f - bf16_as_f32(hi));
}

// Problem constants (from reference setup_inputs)
#define L_SEQ 2048
#define BATCH 32
#define DDIM  512
#define NDIM  1536          // 3*D
#define MROWS (L_SEQ * BATCH)

// ---------------------------------------------------------------------------
// GEMM: X (65536 x 512) @ W (512 x 1536) -> U (65536 x 1536), all fp32 in
// memory, bf16-split WMMA math.  Block tile 128x128, BK=32, 8 waves (4Mx2N),
// wave tile 32x64 (2x4 of 16x16).
// ---------------------------------------------------------------------------
__global__ __launch_bounds__(256) void sru_gemm_bf16split(
    const float* __restrict__ X, const float* __restrict__ W,
    float* __restrict__ U)
{
    __shared__ unsigned short sAhi[128 * 32];   // [m][k]  8 KB
    __shared__ unsigned short sAlo[128 * 32];
    __shared__ unsigned short sBhi[128 * 32];   // transposed: [n][k]
    __shared__ unsigned short sBlo[128 * 32];

    const int tid   = threadIdx.x;
    const int lane  = tid & 31;
    const int wid   = tid >> 5;      // 0..7
    const int waveM = wid >> 1;      // 0..3
    const int waveN = wid & 1;       // 0..1

    const int rowBase = blockIdx.x * 128;   // along M = L*B
    const int colBase = blockIdx.y * 128;   // along N = 3*D

    const v8f vzero = {0.f, 0.f, 0.f, 0.f, 0.f, 0.f, 0.f, 0.f};
    v8f acc[2][4];
    #pragma unroll
    for (int i = 0; i < 2; ++i)
        #pragma unroll
        for (int j = 0; j < 4; ++j) acc[i][j] = vzero;

    // B-tile load mapping: thread -> (kq, nvec): 4 k-rows x 4 n-cols sub-block
    const int nvec = tid & 31;       // n group (4 floats)
    const int kq   = tid >> 5;       // k quad (4 rows)

    // Fragment gather coordinates (wave32 WMMA layouts, CDNA5 ISA)
    const int am  = lane & 15;               // A: row
    const int akb = (lane >> 4) << 3;        // A: K base 0 / 8  (+16 for 2nd half)
    const int bn  = lane & 15;               // B: col
    const int bkb = (lane >> 4) << 3;        // B: K base 0 / 8  (+16 for 2nd half)

    for (int kk = 0; kk < DDIM / 32; ++kk) {
        const int k0 = kk * 32;

        // ---- global loads + fp32 -> bf16 hi/lo split (in registers) ----
        F4 aval[4];
        #pragma unroll
        for (int i = 0; i < 4; ++i) {
            const int s  = tid + i * 256;       // 1024 float4 slots
            const int ar = s >> 3;              // 0..127
            const int av = s & 7;               // float4 within 32 k
            aval[i].v = *(const float4*)(X + (size_t)(rowBase + ar) * DDIM + k0 + av * 4);
        }
        F4 bval[4];
        #pragma unroll
        for (int j = 0; j < 4; ++j)
            bval[j].v = *(const float4*)(W + (size_t)(k0 + kq * 4 + j) * NDIM + colBase + nvec * 4);

        if (kk + 1 < DDIM / 32) {   // warm L2/WGP$ for next k tile
            __builtin_prefetch(X + (size_t)(rowBase + (tid >> 1)) * DDIM + k0 + 32, 0, 1);
            __builtin_prefetch(W + (size_t)(k0 + 32 + kq * 4) * NDIM + colBase + nvec * 4, 0, 1);
        }

        __syncthreads();   // previous iteration's LDS reads complete

        // A -> LDS row-major [m][k]
        #pragma unroll
        for (int i = 0; i < 4; ++i) {
            const int s  = tid + i * 256;
            const int ar = s >> 3;
            const int av = s & 7;
            unsigned short h[4], l[4];
            #pragma unroll
            for (int c = 0; c < 4; ++c) bf16_split(aval[i].f[c], h[c], l[c]);
            const unsigned int h01 = (unsigned int)h[0] | ((unsigned int)h[1] << 16);
            const unsigned int h23 = (unsigned int)h[2] | ((unsigned int)h[3] << 16);
            const unsigned int l01 = (unsigned int)l[0] | ((unsigned int)l[1] << 16);
            const unsigned int l23 = (unsigned int)l[2] | ((unsigned int)l[3] << 16);
            *(uint2*)(sAhi + ar * 32 + av * 4) = make_uint2(h01, h23);
            *(uint2*)(sAlo + ar * 32 + av * 4) = make_uint2(l01, l23);
        }
        // B -> LDS transposed [n][k] so each lane's two runs of 8 consecutive K are contiguous
        #pragma unroll
        for (int c = 0; c < 4; ++c) {
            unsigned short h[4], l[4];
            #pragma unroll
            for (int j = 0; j < 4; ++j) bf16_split(bval[j].f[c], h[j], l[j]);
            const int n = nvec * 4 + c;
            const unsigned int h01 = (unsigned int)h[0] | ((unsigned int)h[1] << 16);
            const unsigned int h23 = (unsigned int)h[2] | ((unsigned int)h[3] << 16);
            const unsigned int l01 = (unsigned int)l[0] | ((unsigned int)l[1] << 16);
            const unsigned int l23 = (unsigned int)l[2] | ((unsigned int)l[3] << 16);
            *(uint2*)(sBhi + n * 32 + kq * 4) = make_uint2(h01, h23);
            *(uint2*)(sBlo + n * 32 + kq * 4) = make_uint2(l01, l23);
        }

        __syncthreads();   // LDS tiles ready

        // ---- fragments + WMMA ----
        FragBF aHi[2], aLo[2];
        #pragma unroll
        for (int tm = 0; tm < 2; ++tm) {
            const int row = waveM * 32 + tm * 16 + am;
            const uint4v* ph = (const uint4v*)(sAhi + row * 32 + akb);
            aHi[tm].q[0] = ph[0];          // K = akb .. akb+7
            aHi[tm].q[1] = ph[2];          // K = 16+akb .. 16+akb+7
            const uint4v* pl = (const uint4v*)(sAlo + row * 32 + akb);
            aLo[tm].q[0] = pl[0];
            aLo[tm].q[1] = pl[2];
        }
        #pragma unroll
        for (int tn = 0; tn < 4; ++tn) {
            const int col = waveN * 64 + tn * 16 + bn;
            FragBF bHi, bLo;
            const uint4v* ph = (const uint4v*)(sBhi + col * 32 + bkb);
            bHi.q[0] = ph[0]; bHi.q[1] = ph[2];
            const uint4v* pl = (const uint4v*)(sBlo + col * 32 + bkb);
            bLo.q[0] = pl[0]; bLo.q[1] = pl[2];
            #pragma unroll
            for (int tm = 0; tm < 2; ++tm) {
                acc[tm][tn] = __builtin_amdgcn_wmma_f32_16x16x32_bf16(
                    false, aHi[tm].bf, false, bHi.bf, (short)0, acc[tm][tn], false, false);
                acc[tm][tn] = __builtin_amdgcn_wmma_f32_16x16x32_bf16(
                    false, aHi[tm].bf, false, bLo.bf, (short)0, acc[tm][tn], false, false);
                acc[tm][tn] = __builtin_amdgcn_wmma_f32_16x16x32_bf16(
                    false, aLo[tm].bf, false, bHi.bf, (short)0, acc[tm][tn], false, false);
            }
        }
    }

    // ---- store C (16x16 f32 layout: VGPR r -> row r or 8+r, lane&15 -> col) ----
    const int mrow = (lane >> 4) << 3;
    const int ncol = lane & 15;
    #pragma unroll
    for (int tm = 0; tm < 2; ++tm)
        #pragma unroll
        for (int tn = 0; tn < 4; ++tn) {
            const int row0 = rowBase + waveM * 32 + tm * 16 + mrow;
            const int col  = colBase + waveN * 64 + tn * 16 + ncol;
            float* p = U + (size_t)row0 * NDIM + col;
            #pragma unroll
            for (int r = 0; r < 8; ++r) p[(size_t)r * NDIM] = acc[tm][tn][r];
        }
}

// ---------------------------------------------------------------------------
// SRU scan: 1 thread per (b, d); sequential over L.
// U columns for feature d are 3d+{0,1,2} (contiguous 12 bytes).
// ---------------------------------------------------------------------------
__device__ __forceinline__ float sigmoidf_fast(float z) {
    return 1.0f / (1.0f + __expf(-z));
}

__global__ __launch_bounds__(256) void sru_scan(
    const float* __restrict__ x, const float* __restrict__ U,
    const float* __restrict__ wc, const float* __restrict__ bias,
    const float* __restrict__ c_init, float* __restrict__ h_out,
    float* __restrict__ c_out)
{
    const int idx = blockIdx.x * 256 + threadIdx.x;   // 0 .. B*D-1
    const int d = idx & (DDIM - 1);
    const int b = idx >> 9;

    const float fw = wc[d];
    const float rw = wc[DDIM + d];
    const float fb = bias[d];
    const float rb = bias[DDIM + d];

    float c = c_init[idx];

    #pragma unroll 2
    for (int t = 0; t < L_SEQ; ++t) {
        const size_t row = (size_t)t * BATCH + b;
        const float* u = U + row * NDIM + d * 3;
        const float u0 = u[0];
        const float u1 = u[1];
        const float u2 = u[2];
        const float xt = x[row * DDIM + d];

        const float f = sigmoidf_fast(u1 + fb + c * fw);
        const float r = sigmoidf_fast(u2 + rb + c * rw);
        c = u0 + (c - u0) * f;
        const float h = xt + (c - xt) * r;
        h_out[row * DDIM + d] = h;
    }
    c_out[idx] = c;
}

// ---------------------------------------------------------------------------
extern "C" void kernel_launch(void* const* d_in, const int* in_sizes, int n_in,
                              void* d_out, int out_size, void* d_ws, size_t ws_size,
                              hipStream_t stream)
{
    const float* x      = (const float*)d_in[0];   // (L, B, D)
    const float* weight = (const float*)d_in[1];   // (D, 3D)
    const float* wc     = (const float*)d_in[2];   // (2D,)
    const float* bias   = (const float*)d_in[3];   // (2D,)
    const float* c_init = (const float*)d_in[4];   // (B, D)

    float* U     = (float*)d_ws;                   // (L*B, 3D) fp32, 403 MB
    float* h_out = (float*)d_out;                  // (L, B, D)
    float* c_out = h_out + (size_t)L_SEQ * BATCH * DDIM;

    dim3 ggrid(MROWS / 128, NDIM / 128, 1);        // 512 x 12
    sru_gemm_bf16split<<<ggrid, dim3(256, 1, 1), 0, stream>>>(x, weight, U);

    dim3 sgrid((BATCH * DDIM) / 256, 1, 1);        // 64 blocks
    sru_scan<<<sgrid, dim3(256, 1, 1), 0, stream>>>(x, U, wc, bias, c_init, h_out, c_out);
}
